// MambaEncoder_25812753449329
// MI455X (gfx1250) — compile-verified, matrix operand order checked by reading, not run on hardware
//
#include <hip/hip_runtime.h>
#include <hip/hip_bf16.h>

// ---------------------------------------------------------------------------
// 2-layer GRU over 128 ragged (sorted-by-graph) sequences + masked mean.
// CDNA5 (gfx1250, wave32) implementation:
//   K1 seg_offsets : per-graph start/count via binary search on sorted batch
//   K2 cvt_f16     : f32 -> f16 copies of x and the 4 weight matrices
//   K3 gemm_gx0    : gx0 = x @ W_ih0^T  (65536x256 @ 256x768), v_wmma f16->f32
//   K4 gru_scan    : 8 persistent WGs x 16 graphs; per step 3 fused WMMA GEMMs
//                    (gh0, gx1, gh1) with h-state in LDS, gates in f32.
// ---------------------------------------------------------------------------

typedef __attribute__((ext_vector_type(16))) _Float16 v16h;
typedef __attribute__((ext_vector_type(8)))  float    v8f;

#define NN   65536   // nodes
#define DIM  256     // input dim == hidden
#define H    256
#define G3   768     // 3*H gate rows
#define NG   128     // graphs
#define GPB  16      // graphs per scan workgroup
#define SCAN_BLOCKS (NG / GPB)          // 8
#define SCAN_THREADS 512                // 16 waves
#define SCAN_WAVES  16
#define NT_PER_WAVE 3                   // 48 N-tiles / 16 waves

// LDS carve (bytes) for gru_scan dynamic shared memory
#define OFF_BUFA 0                      // 16x768 f32 = 49152
#define OFF_BUFB 49152                  // 16x768 f32 = 49152
#define OFF_H0   98304                  // 16x256 f16 = 8192
#define OFF_H1   106496                 // 16x256 f16 = 8192
#define OFF_META 114688                 // ints: lens[16], starts[16], maxlen
#define SCAN_LDS_BYTES 114944

// ---------------------------------------------------------------- K1
__global__ void seg_offsets(const int* __restrict__ batch,
                            int* __restrict__ starts, int* __restrict__ counts) {
    int g = threadIdx.x;                     // 128 threads, 1 block
    if (g >= NG) return;
    int lo = 0, hi = NN;
    while (lo < hi) { int m = (lo + hi) >> 1; if (batch[m] < g) lo = m + 1; else hi = m; }
    int s = lo;
    lo = s; hi = NN;
    while (lo < hi) { int m = (lo + hi) >> 1; if (batch[m] < g + 1) lo = m + 1; else hi = m; }
    starts[g] = s;
    counts[g] = lo - s;
}

// ---------------------------------------------------------------- K2
__global__ void cvt_f16(const float* __restrict__ src, _Float16* __restrict__ dst, int n) {
    int i = blockIdx.x * blockDim.x + threadIdx.x;
    if (i < n) dst[i] = (_Float16)src[i];
}

// ---------------------------------------------------------------- K3
// gx0[n, 0:768] = x[n, :] @ W_ih0^T   (bias added later in the scan)
// Block = 256 threads (8 waves); block b owns rows [16b, 16b+16); wave w owns
// columns [96w, 96w+96) as six 16x16 tiles, K=256 -> 8 chained WMMAs per tile.
__global__ __launch_bounds__(256) void gemm_gx0(const _Float16* __restrict__ xh,
                                                const _Float16* __restrict__ W,
                                                _Float16* __restrict__ gx0) {
    const int lane = threadIdx.x & 31;
    const int wave = threadIdx.x >> 5;       // 0..7
    const int m0   = blockIdx.x * 16;        // 4096 blocks

    // A fragments: lane L holds row m0+(L&15), k = (L>>4)*16 .. +15 (32B/lane)
    v16h a[8];
    const _Float16* abase = xh + (size_t)(m0 + (lane & 15)) * DIM + ((lane >> 4) << 4);
#pragma unroll
    for (int kk = 0; kk < 8; ++kk) a[kk] = *(const v16h*)(abase + kk * 32);

#pragma unroll
    for (int c = 0; c < 6; ++c) {
        const int n0 = wave * 96 + c * 16;
        // B fragments: lane L holds W row n0+(L&15) (== B column), contiguous k
        const _Float16* bbase = W + (size_t)(n0 + (lane & 15)) * DIM + ((lane >> 4) << 4);
        v8f acc = {};
#pragma unroll
        for (int kk = 0; kk < 8; ++kk) {
            v16h b = *(const v16h*)(bbase + kk * 32);
            acc = __builtin_amdgcn_wmma_f32_16x16x32_f16(false, a[kk], false, b,
                                                         (short)0, acc, false, false);
        }
        const int n  = n0 + (lane & 15);
        const int mb = (lane >> 4) * 8;
#pragma unroll
        for (int i = 0; i < 8; ++i)
            gx0[(size_t)(m0 + mb + i) * G3 + n] = (_Float16)acc[i];
    }
}

// One [16,256] @ [256,768] step: hA (LDS, f16 row-major stride 256) times W^T
// (global f16 [768,256] row-major), f32 result into obuf (LDS, stride 768).
__device__ __forceinline__ void gemm_step(const _Float16* __restrict__ hA,
                                          const _Float16* __restrict__ W,
                                          float* __restrict__ obuf,
                                          int lane, int wave) {
    v16h a[8];
    const _Float16* abase = hA + (lane & 15) * H + ((lane >> 4) << 4);
#pragma unroll
    for (int kk = 0; kk < 8; ++kk) a[kk] = *(const v16h*)(abase + kk * 32);
#pragma unroll
    for (int c = 0; c < NT_PER_WAVE; ++c) {
        const int n0 = wave * (16 * NT_PER_WAVE) + c * 16;     // 16 waves * 48 = 768
        const _Float16* bbase = W + (size_t)(n0 + (lane & 15)) * H + ((lane >> 4) << 4);
        v8f acc = {};
#pragma unroll
        for (int kk = 0; kk < 8; ++kk) {
            v16h b = *(const v16h*)(bbase + kk * 32);
            acc = __builtin_amdgcn_wmma_f32_16x16x32_f16(false, a[kk], false, b,
                                                         (short)0, acc, false, false);
        }
        const int n  = n0 + (lane & 15);
        const int mb = (lane >> 4) * 8;
#pragma unroll
        for (int i = 0; i < 8; ++i)
            obuf[(mb + i) * G3 + n] = acc[i];
    }
}

__device__ __forceinline__ float sigm(float v) { return 1.f / (1.f + __expf(-v)); }

// ---------------------------------------------------------------- K4
__global__ __launch_bounds__(SCAN_THREADS) void gru_scan(
    const _Float16* __restrict__ gx0,
    const _Float16* __restrict__ Whh0, const _Float16* __restrict__ Wih1,
    const _Float16* __restrict__ Whh1,
    const float* __restrict__ b_ih0, const float* __restrict__ b_hh0,
    const float* __restrict__ b_ih1, const float* __restrict__ b_hh1,
    const int* __restrict__ starts, const int* __restrict__ counts,
    float* __restrict__ out)
{
    extern __shared__ char smem[];
    float*    bufA = (float*)(smem + OFF_BUFA);     // 16x768 gate staging (f32)
    float*    bufB = (float*)(smem + OFF_BUFB);     // 16x768 gate staging (f32)
    _Float16* h0   = (_Float16*)(smem + OFF_H0);    // 16x256 layer-0 state
    _Float16* h1   = (_Float16*)(smem + OFF_H1);    // 16x256 layer-1 state
    int*      meta = (int*)(smem + OFF_META);       // [0:16) lens, [16:32) starts, [32] maxlen

    const int tid  = threadIdx.x;
    const int lane = tid & 31;
    const int wave = tid >> 5;
    const int gbase = blockIdx.x * GPB;             // 8 blocks x 16 graphs

    for (int i = tid; i < GPB * H; i += SCAN_THREADS) {
        h0[i] = (_Float16)0.f;
        h1[i] = (_Float16)0.f;
    }
    if (tid < GPB) {
        meta[tid]      = counts[gbase + tid];
        meta[16 + tid] = starts[gbase + tid];
    }
    __syncthreads();
    if (tid == 0) {
        int m = 0;
#pragma unroll
        for (int g = 0; g < GPB; ++g) m = (meta[g] > m) ? meta[g] : m;
        meta[32] = m;
    }
    __syncthreads();
    const int maxLen = meta[32];

    // Elementwise ownership: thread (gh, j) owns column j of 8 graphs.
    const int j  = tid & 255;
    const int gh = tid >> 8;                        // 0/1 -> graphs gh*8 .. gh*8+7
    float sum[8];
#pragma unroll
    for (int i = 0; i < 8; ++i) sum[i] = 0.f;
    const float bir0 = b_ih0[j], biz0 = b_ih0[256 + j], bin0 = b_ih0[512 + j];
    const float bhr0 = b_hh0[j], bhz0 = b_hh0[256 + j], bhn0 = b_hh0[512 + j];
    const float bir1 = b_ih1[j], biz1 = b_ih1[256 + j], bin1 = b_ih1[512 + j];
    const float bhr1 = b_hh1[j], bhz1 = b_hh1[256 + j], bhn1 = b_hh1[512 + j];

    for (int t = 0; t < maxLen; ++t) {
        // ---- gh0 = h0 @ W_hh0^T -> bufA ----
        gemm_step(h0, Whh0, bufA, lane, wave);
        __syncthreads();

        // ---- layer-0 gates ----
#pragma unroll
        for (int gi = 0; gi < 8; ++gi) {
            const int g   = gh * 8 + gi;
            const int len = meta[g];
            const int st  = meta[16 + g];
            const bool valid = (t < len);
            const int node   = st + (valid ? t : 0);     // clamped, masked below
            const _Float16* gp = gx0 + (size_t)node * G3;
            const float xr = (float)gp[j]       + bir0;
            const float xz = (float)gp[256 + j] + biz0;
            const float xn = (float)gp[512 + j] + bin0;
            const float hr = bufA[g * G3 + j]       + bhr0;
            const float hz = bufA[g * G3 + 256 + j] + bhz0;
            const float hn = bufA[g * G3 + 512 + j] + bhn0;
            const float r = sigm(xr + hr);
            const float z = sigm(xz + hz);
            const float n = tanhf(xn + r * hn);
            const float hold = (float)h0[g * H + j];
            const float hnew = (1.f - z) * n + z * hold;
            h0[g * H + j] = (_Float16)(valid ? hnew : hold);
        }
        __syncthreads();

        // ---- gx1 = h0' @ W_ih1^T -> bufA ; gh1 = h1 @ W_hh1^T -> bufB ----
        gemm_step(h0, Wih1, bufA, lane, wave);
        gemm_step(h1, Whh1, bufB, lane, wave);
        __syncthreads();

        // ---- layer-1 gates + masked-mean accumulation ----
#pragma unroll
        for (int gi = 0; gi < 8; ++gi) {
            const int g   = gh * 8 + gi;
            const int len = meta[g];
            const bool valid = (t < len);
            const float xr = bufA[g * G3 + j]       + bir1;
            const float xz = bufA[g * G3 + 256 + j] + biz1;
            const float xn = bufA[g * G3 + 512 + j] + bin1;
            const float hr = bufB[g * G3 + j]       + bhr1;
            const float hz = bufB[g * G3 + 256 + j] + bhz1;
            const float hn = bufB[g * G3 + 512 + j] + bhn1;
            const float r = sigm(xr + hr);
            const float z = sigm(xz + hz);
            const float n = tanhf(xn + r * hn);
            const float hold = (float)h1[g * H + j];
            const float hnew = (1.f - z) * n + z * hold;
            const float hkeep = valid ? hnew : hold;
            h1[g * H + j] = (_Float16)hkeep;
            sum[gi] += valid ? hkeep : 0.f;
        }
        __syncthreads();
    }

#pragma unroll
    for (int gi = 0; gi < 8; ++gi) {
        const int g   = gh * 8 + gi;
        const int cnt = meta[g];
        const float denom = (float)(cnt > 0 ? cnt : 1);
        out[(size_t)(gbase + g) * H + j] = sum[gi] / denom;
    }
}

// ---------------------------------------------------------------------------
extern "C" void kernel_launch(void* const* d_in, const int* in_sizes, int n_in,
                              void* d_out, int out_size, void* d_ws, size_t ws_size,
                              hipStream_t stream) {
    const float* x     = (const float*)d_in[0];
    const int*   batch = (const int*)d_in[1];      // int32 on device (JAX x64 off)
    const float* W_ih0 = (const float*)d_in[2];
    const float* W_hh0 = (const float*)d_in[3];
    const float* b_ih0 = (const float*)d_in[4];
    const float* b_hh0 = (const float*)d_in[5];
    const float* W_ih1 = (const float*)d_in[6];
    const float* W_hh1 = (const float*)d_in[7];
    const float* b_ih1 = (const float*)d_in[8];
    const float* b_hh1 = (const float*)d_in[9];
    float* out = (float*)d_out;

    // Workspace layout (~136 MB): meta | 4x f16 weights | f16 x | f16 gx0
    char* ws = (char*)d_ws;
    int* starts = (int*)ws;                         // 128
    int* counts = starts + 128;                     // 128
    _Float16* Wih0h = (_Float16*)(ws + 1024);
    _Float16* Whh0h = Wih0h + (size_t)G3 * DIM;
    _Float16* Wih1h = Whh0h + (size_t)G3 * DIM;
    _Float16* Whh1h = Wih1h + (size_t)G3 * DIM;
    _Float16* xh    = Whh1h + (size_t)G3 * DIM;
    _Float16* gx0   = xh + (size_t)NN * DIM;

    seg_offsets<<<1, 128, 0, stream>>>(batch, starts, counts);

    const int wN = G3 * DIM;                        // 196608
    cvt_f16<<<(wN + 255) / 256, 256, 0, stream>>>(W_ih0, Wih0h, wN);
    cvt_f16<<<(wN + 255) / 256, 256, 0, stream>>>(W_hh0, Whh0h, wN);
    cvt_f16<<<(wN + 255) / 256, 256, 0, stream>>>(W_ih1, Wih1h, wN);
    cvt_f16<<<(wN + 255) / 256, 256, 0, stream>>>(W_hh1, Whh1h, wN);
    const int xN = NN * DIM;                        // 16M
    cvt_f16<<<(xN + 255) / 256, 256, 0, stream>>>(x, xh, xN);

    gemm_gx0<<<NN / 16, 256, 0, stream>>>(xh, Wih0h, gx0);

    gru_scan<<<SCAN_BLOCKS, SCAN_THREADS, SCAN_LDS_BYTES, stream>>>(
        gx0, Whh0h, Wih1h, Whh1h,
        b_ih0, b_hh0, b_ih1, b_hh1,
        starts, counts, out);
}
